// AttentionHead_53979148976304
// MI455X (gfx1250) — matrix operand order repaired and checked by reading, not compiled, not run on hardware
//
#include <hip/hip_runtime.h>
#include <hip/hip_bf16.h>

typedef __attribute__((ext_vector_type(16))) _Float16 v16h;
typedef __attribute__((ext_vector_type(8)))  _Float16 v8h;
typedef __attribute__((ext_vector_type(8)))  float    v8f;

#define EMB   768
#define HEADD 96
#define SEQ   2048
#define BATCH 16

// ---------------------------------------------------------------------------
// gfx1250 async global->LDS copy (VGLOBAL GLOBAL_LOAD_ASYNC_TO_LDS_B128,
// tracked by ASYNCcnt). VDST = LDS byte offset, VADDR = 64-bit address.
// ---------------------------------------------------------------------------
__device__ __forceinline__ void async_load_b128(unsigned lds_off, const void* gaddr) {
    asm volatile("global_load_async_to_lds_b128 %0, %1, off"
                 :: "v"(lds_off), "v"(gaddr)
                 : "memory");
}
__device__ __forceinline__ void wait_asynccnt0() {
    asm volatile("s_wait_asynccnt 0" ::: "memory");
}

// ---------------------------------------------------------------------------
// Kernel 0: convert Wq/Wk/Wv [768][96] f32 -> Wt[3][96][768] f16 (transposed)
// so the projection's WMMA B-operand is two contiguous 16-byte loads per lane.
// ---------------------------------------------------------------------------
__global__ void __launch_bounds__(256)
wt_convert_kernel(const float* __restrict__ Wq,
                  const float* __restrict__ Wk,
                  const float* __restrict__ Wv,
                  _Float16* __restrict__ wt) {
    const int idx = blockIdx.x * 256 + threadIdx.x;   // over 3*96*768
    const int which = idx / (HEADD * EMB);
    const int rem   = idx % (HEADD * EMB);
    const int h     = rem / EMB;
    const int e     = rem % EMB;
    const float* W = (which == 0) ? Wq : (which == 1) ? Wk : Wv;
    wt[idx] = (_Float16)W[(size_t)e * HEADD + h];
}

// ---------------------------------------------------------------------------
// Kernel 1: fused QKV projection, f32 x -> f16 q/k/vt, f32 accumulate WMMA.
// Grid: (2048 m-tiles, 18) where y = which(0..2)*6 + ntile(0..5). Block: 32.
// q,k stored row-major [B*S, 96]; v stored transposed per batch [96, S].
// ---------------------------------------------------------------------------
__global__ void __launch_bounds__(32)
qkv_proj_kernel(const float* __restrict__ x,
                const _Float16* __restrict__ wt,
                _Float16* __restrict__ qf,
                _Float16* __restrict__ kf,
                _Float16* __restrict__ vtf) {
    const int lane = threadIdx.x & 31;
    const int n    = lane & 15;     // col within tile / A row
    const int g    = lane >> 4;     // lane half
    const int mt   = blockIdx.x;
    const int which = blockIdx.y / 6;          // 0=q 1=k 2=v
    const int nc    = (blockIdx.y % 6) * 16;   // head-dim tile base
    const int mr0 = mt * 16;
    const float*    xrow = x  + (size_t)(mr0 + n) * EMB;
    const _Float16* wrow = wt + (size_t)which * HEADD * EMB + (size_t)(nc + n) * EMB;

    v8f acc = {};
    for (int kc = 0; kc < EMB; kc += 32) {
        // A fragment (16x32 f16): halves 0..7 = K=g*8+0..7, 8..15 = K=16+g*8+0..7
        v16h a;
        const float4 x0 = *(const float4*)(xrow + kc + g * 8);
        const float4 x1 = *(const float4*)(xrow + kc + g * 8 + 4);
        const float4 x2 = *(const float4*)(xrow + kc + 16 + g * 8);
        const float4 x3 = *(const float4*)(xrow + kc + 16 + g * 8 + 4);
        a[0]  = (_Float16)x0.x; a[1]  = (_Float16)x0.y; a[2]  = (_Float16)x0.z; a[3]  = (_Float16)x0.w;
        a[4]  = (_Float16)x1.x; a[5]  = (_Float16)x1.y; a[6]  = (_Float16)x1.z; a[7]  = (_Float16)x1.w;
        a[8]  = (_Float16)x2.x; a[9]  = (_Float16)x2.y; a[10] = (_Float16)x2.z; a[11] = (_Float16)x2.w;
        a[12] = (_Float16)x3.x; a[13] = (_Float16)x3.y; a[14] = (_Float16)x3.z; a[15] = (_Float16)x3.w;
        // B fragment (32x16 f16): lane n = col nc+n, halves 0..7 = K=g*8+0..7, 8..15 = K=16+g*8+0..7
        v16h b;
        {
            v8h blo = *(const v8h*)(wrow + kc + g * 8);
            v8h bhi = *(const v8h*)(wrow + kc + 16 + g * 8);
            for (int j = 0; j < 8; ++j) { b[j] = blo[j]; b[j + 8] = bhi[j]; }
        }
        acc = __builtin_amdgcn_wmma_f32_16x16x32_f16(false, a, false, b,
                                                     (short)0, acc, false, false);
    }

    if (which == 2) {
        // transposed store: vt[b][nc+n][s0 + 8g + v], contiguous 8 halves
        const int bb = mr0 / SEQ;
        const int s0 = (mr0 % SEQ) + 8 * g;
        v8h hv;
#pragma unroll
        for (int v = 0; v < 8; ++v) hv[v] = (_Float16)acc[v];
        *(v8h*)(vtf + ((size_t)bb * HEADD + nc + n) * SEQ + s0) = hv;
    } else {
        _Float16* dst = (which == 0) ? qf : kf;
#pragma unroll
        for (int v = 0; v < 8; ++v)
            dst[(size_t)(mr0 + v + 8 * g) * HEADD + nc + n] = (_Float16)acc[v];
    }
}

// ---------------------------------------------------------------------------
// Kernel 2: causal flash attention, FA-2 decomposition, double-buffered.
// Block = 4 waves = 64 query rows; block-uniform loop over 32-key blocks.
// K tile and V^T tile ping-pong staged into LDS with async b128 copies
// (ASYNCcnt) that run underneath the current block's WMMA + softmax.
// Grid: (SEQ/64, BATCH), block 128.
// ---------------------------------------------------------------------------
__global__ void __launch_bounds__(128)
flash_attn_kernel(const _Float16* __restrict__ qf,
                  const _Float16* __restrict__ kf,
                  const _Float16* __restrict__ vtf,
                  float* __restrict__ out) {
    __shared__ __align__(32) _Float16 ldsK[2][32 * HEADD];  // [buf][key][96]  2x6144 B
    __shared__ __align__(32) _Float16 ldsV[2][HEADD * 32];  // [buf][head][32] 2x6144 B
    __shared__ __align__(16) _Float16 lds_p[4][16][32];     // P transpose, 4 KB

    const int tid  = threadIdx.x;
    const int lane = tid & 31;
    const int wid  = tid >> 5;
    const int n    = lane & 15;
    const int g    = lane >> 4;
    const int b    = blockIdx.y;
    const int qb0  = blockIdx.x * 64;        // block's query base
    const int qr0  = qb0 + wid * 16;         // wave's query tile

    const _Float16* qb = qf  + (size_t)b * SEQ * HEADD;
    const _Float16* kb = kf  + (size_t)b * SEQ * HEADD;
    const _Float16* vb = vtf + (size_t)b * HEADD * SEQ;

    const unsigned ldsK_base = (unsigned)(size_t)&ldsK[0][0];
    const unsigned ldsV_base = (unsigned)(size_t)&ldsV[0][0];
    const unsigned KBUF = 32 * HEADD * 2;    // bytes per K buffer
    const unsigned VBUF = HEADD * 32 * 2;    // bytes per V buffer

    // async-stage one 32-key block (K tile [32][96], V^T tile [96][32])
    auto stage = [&](int bsel, int kr) {
#pragma unroll
        for (int i = 0; i < 3; ++i) {
            const int id  = tid * 3 + i;            // 0..383 (16B chunks)
            const int row = id / 12, c = id % 12;   // 12 chunks per 192B K row
            async_load_b128(ldsK_base + bsel * KBUF + id * 16,
                            kb + (size_t)(kr + row) * HEADD + c * 8);
        }
#pragma unroll
        for (int i = 0; i < 3; ++i) {
            const int id  = tid * 3 + i;            // 0..383
            const int row = id / 4, c = id % 4;     // 4 chunks per 64B V^T row
            async_load_b128(ldsV_base + bsel * VBUF + id * 16,
                            vb + (size_t)row * SEQ + kr + c * 8);
        }
    };

    // Q A-fragments: 3 chunks of 16x32 (head dim 96)
    v16h aq[3];
    const _Float16* qrow = qb + (size_t)(qr0 + n) * HEADD;
#pragma unroll
    for (int c = 0; c < 3; ++c) {
        v8h lo = *(const v8h*)(qrow + c * 32 + g * 8);
        v8h hi = *(const v8h*)(qrow + c * 32 + 16 + g * 8);
#pragma unroll
        for (int j = 0; j < 8; ++j) { aq[c][j] = lo[j]; aq[c][j + 8] = hi[j]; }
    }

    const float scale = 0.10206207261596575f;  // 96^-0.5
    float m_i[8], l_i[8];
    v8f o[6];
#pragma unroll
    for (int v = 0; v < 8; ++v) { m_i[v] = -1e30f; l_i[v] = 0.0f; }
#pragma unroll
    for (int t = 0; t < 6; ++t) o[t] = (v8f){};

    const int kr_end = qb0 + 64;   // exclusive, block-uniform
    stage(0, 0);                   // prologue: fill buffer 0
    int buf = 0;

    for (int kr0 = 0; kr0 < kr_end; kr0 += 32) {
        wait_asynccnt0();          // this wave's copies for buf have landed
        __syncthreads();           // everyone done with the buffer we overwrite next

        const int kn = kr0 + 32;
        if (kn < kr_end) stage(buf ^ 1, kn);   // copy next block under compute
        // warm L2 two blocks ahead (one ahead is already in the async engine)
        const int kp = kr0 + 64;
        if (kp < kr_end) {
            __builtin_prefetch(kb + (size_t)(kp + (tid & 31)) * HEADD, 0, 2);
            if (tid < HEADD) __builtin_prefetch(vb + (size_t)tid * SEQ + kp, 0, 2);
        }

        if (kr0 <= qr0 + 15) {     // wave-uniform causal bound; EXEC stays full
            const bool has2 = (kr0 + 16 <= qr0 + 15);
            const _Float16* Kc = &ldsK[buf][0];
            const _Float16* Vc = &ldsV[buf][0];

            // ---- scores from LDS K tile ----
            v8f s0 = {}, s1 = {};
#pragma unroll
            for (int c = 0; c < 3; ++c) {
                v16h bk0;
                {
                    v8h blo = *(const v8h*)(Kc + n * HEADD + c * 32 + g * 8);
                    v8h bhi = *(const v8h*)(Kc + n * HEADD + c * 32 + 16 + g * 8);
                    for (int j = 0; j < 8; ++j) { bk0[j] = blo[j]; bk0[j + 8] = bhi[j]; }
                }
                s0 = __builtin_amdgcn_wmma_f32_16x16x32_f16(false, aq[c], false, bk0,
                                                            (short)0, s0, false, false);
            }
            if (has2) {
#pragma unroll
                for (int c = 0; c < 3; ++c) {
                    v16h bk1;
                    {
                        v8h blo = *(const v8h*)(Kc + (16 + n) * HEADD + c * 32 + g * 8);
                        v8h bhi = *(const v8h*)(Kc + (16 + n) * HEADD + c * 32 + 16 + g * 8);
                        for (int j = 0; j < 8; ++j) { bk1[j] = blo[j]; bk1[j + 8] = bhi[j]; }
                    }
                    s1 = __builtin_amdgcn_wmma_f32_16x16x32_f16(false, aq[c], false, bk1,
                                                                (short)0, s1, false, false);
                }
            }

            // ---- scale + causal mask + block row max ----
            float mb[8];
#pragma unroll
            for (int v = 0; v < 8; ++v) {
                const int qrow_g = qr0 + v + 8 * g;
                s0[v] = (kr0 + n <= qrow_g) ? s0[v] * scale : -1e30f;
                mb[v] = s0[v];
            }
            if (has2) {
#pragma unroll
                for (int v = 0; v < 8; ++v) {
                    const int qrow_g = qr0 + v + 8 * g;
                    s1[v] = (kr0 + 16 + n <= qrow_g) ? s1[v] * scale : -1e30f;
                    mb[v] = fmaxf(mb[v], s1[v]);
                }
            }
#pragma unroll
            for (int off = 1; off < 16; off <<= 1)
#pragma unroll
                for (int v = 0; v < 8; ++v)
                    mb[v] = fmaxf(mb[v], __shfl_xor(mb[v], off, 32));

            // ---- online softmax update ----
            float alpha[8], rs[8];
            v8f p0, p1;
#pragma unroll
            for (int v = 0; v < 8; ++v) {
                const float mnew = fmaxf(m_i[v], mb[v]);
                alpha[v] = __expf(m_i[v] - mnew);
                m_i[v] = mnew;
                p0[v] = __expf(s0[v] - mnew);
                p1[v] = has2 ? __expf(s1[v] - mnew) : 0.0f;
                rs[v] = p0[v] + p1[v];
            }
#pragma unroll
            for (int off = 1; off < 16; off <<= 1)
#pragma unroll
                for (int v = 0; v < 8; ++v)
                    rs[v] += __shfl_xor(rs[v], off, 32);
#pragma unroll
            for (int v = 0; v < 8; ++v) l_i[v] = l_i[v] * alpha[v] + rs[v];
#pragma unroll
            for (int t = 0; t < 6; ++t)
#pragma unroll
                for (int v = 0; v < 8; ++v) o[t][v] *= alpha[v];

            // ---- P: D-layout -> A-layout transpose via per-wave LDS ----
            _Float16* lp = &lds_p[wid][0][0];
#pragma unroll
            for (int v = 0; v < 8; ++v) {
                lp[(v + 8 * g) * 32 + n]      = (_Float16)p0[v];
                lp[(v + 8 * g) * 32 + n + 16] = (_Float16)p1[v];
            }
            asm volatile("s_wait_dscnt 0" ::: "memory");
            v16h ap;
            {
                v8h lo = *(const v8h*)(lp + n * 32 + g * 8);
                v8h hi = *(const v8h*)(lp + n * 32 + 16 + g * 8);
#pragma unroll
                for (int j = 0; j < 8; ++j) { ap[j] = lo[j]; ap[j + 8] = hi[j]; }
            }
            asm volatile("s_wait_dscnt 0" ::: "memory");

            // ---- O += P @ V from LDS V^T tile ----
#pragma unroll
            for (int t = 0; t < 6; ++t) {
                v16h bv;
                {
                    v8h blo = *(const v8h*)(Vc + (t * 16 + n) * 32 + g * 8);
                    v8h bhi = *(const v8h*)(Vc + (t * 16 + n) * 32 + 16 + g * 8);
                    for (int j = 0; j < 8; ++j) { bv[j] = blo[j]; bv[j + 8] = bhi[j]; }
                }
                o[t] = __builtin_amdgcn_wmma_f32_16x16x32_f16(false, ap, false, bv,
                                                              (short)0, o[t], false, false);
            }
        }
        buf ^= 1;
    }

    // ---- finalize: divide by row sums, store f32 ----
    float invl[8];
#pragma unroll
    for (int v = 0; v < 8; ++v) invl[v] = 1.0f / l_i[v];
#pragma unroll
    for (int t = 0; t < 6; ++t)
#pragma unroll
        for (int v = 0; v < 8; ++v)
            out[((size_t)b * SEQ + qr0 + v + 8 * g) * HEADD + t * 16 + n] =
                o[t][v] * invl[v];
}

// ---------------------------------------------------------------------------
extern "C" void kernel_launch(void* const* d_in, const int* in_sizes, int n_in,
                              void* d_out, int out_size, void* d_ws, size_t ws_size,
                              hipStream_t stream) {
    const float* x  = (const float*)d_in[0];
    const float* Wq = (const float*)d_in[1];
    const float* Wk = (const float*)d_in[2];
    const float* Wv = (const float*)d_in[3];
    float* out = (float*)d_out;

    const size_t elems = (size_t)BATCH * SEQ * HEADD;     // 3,145,728 halves each
    _Float16* qf  = (_Float16*)d_ws;
    _Float16* kf  = qf + elems;
    _Float16* vtf = kf + elems;
    _Float16* wtf = vtf + elems;                          // 3*96*768 halves

    // Weight convert+transpose: 3*96*768 = 221184 elements
    wt_convert_kernel<<<(3 * HEADD * EMB) / 256, 256, 0, stream>>>(Wq, Wk, Wv, wtf);

    // QKV projection: 2048 m-tiles x (3 matrices * 6 head tiles), 1 wave each
    dim3 pg(SEQ * BATCH / 16, 18);
    qkv_proj_kernel<<<pg, 32, 0, stream>>>(x, wtf, qf, kf, vtf);

    // Flash attention: one block per 64 query rows, per batch
    dim3 fg(SEQ / 64, BATCH);
    flash_attn_kernel<<<fg, 128, 0, stream>>>(qf, kf, vtf, out);
}
